// CudaFastWeightSumQueryTwoLinearTransformerLayer_15891378995459
// MI455X (gfx1250) — matrix operand order repaired and checked by reading, not compiled, not run on hardware
//
#include <hip/hip_runtime.h>

// ---------------------------------------------------------------------------
// Types for CDNA5 WMMA (wave32): v_wmma_f32_16x16x32_bf16
// ---------------------------------------------------------------------------
typedef __attribute__((ext_vector_type(16))) __bf16 v16bf;
typedef __attribute__((ext_vector_type(8)))  __bf16 v8bf;
typedef __attribute__((ext_vector_type(8)))  float  v8f;

// CDNA5 async global->LDS path (ASYNCcnt), guarded so compile never regresses.
#if defined(__gfx1250__) && __has_builtin(__builtin_amdgcn_global_load_async_to_lds_b128) && __has_builtin(__builtin_amdgcn_s_wait_asynccnt)
#define USE_ASYNC 1
#else
#define USE_ASYNC 0
#endif

static __device__ __forceinline__ v8f vzero8() {
  v8f z = {0.f, 0.f, 0.f, 0.f, 0.f, 0.f, 0.f, 0.f};
  return z;
}

static __device__ __forceinline__ v8f wmma_bf16(v16bf a, v16bf b, v8f c) {
  // (neg_a, A, neg_b, B, c_mod, C, reuse_a, reuse_b)
  return __builtin_amdgcn_wmma_f32_16x16x32_bf16(false, a, false, b, (short)0, c,
                                                 false, false);
}

#if USE_ASYNC
// Builtin signature (from hipcc diagnostic): pointee is gcc-style int vector_size(16)
typedef int b128_t __attribute__((vector_size(16)));
static __device__ __forceinline__ void async_copy_b128(const __bf16* g, __bf16* l) {
  __builtin_amdgcn_global_load_async_to_lds_b128(
      (__attribute__((address_space(1))) b128_t*)g,
      (__attribute__((address_space(3))) b128_t*)l, 0, 0);
}
#endif

// A fragment: 16x32 (MxK) bf16, source row-major [*, lda].
// lane layout (ISA 7.12.2): lanes 0-15 row M=lane, K={0..7,16..23};
// lanes 16-31 row M=lane-16, K={8..15,24..31}. Two contiguous 8-elem runs.
static __device__ __forceinline__ v16bf load_a_frag(const __bf16* __restrict__ src,
                                                    int lda, int row0, int k0,
                                                    int lane) {
  int r  = row0 + (lane & 15);
  int kb = k0 + ((lane >> 4) << 3);
  const __bf16* p = src + (size_t)r * lda + kb;
  v8bf lo = *(const v8bf*)(p);
  v8bf hi = *(const v8bf*)(p + 16);
  return __builtin_shufflevector(lo, hi, 0, 1, 2, 3, 4, 5, 6, 7,
                                 8, 9, 10, 11, 12, 13, 14, 15);
}

// B fragment: 32x16 (KxN) bf16 where B[k][n] = src[n][k] (src row-major [N,ldb]).
// lanes 0-15: col N=lane, K={0..7,16..23}; lanes 16-31: col N=lane-16, K={8..15,24..31}.
// Two contiguous 8-elem runs per lane.
static __device__ __forceinline__ v16bf load_bT_frag(const __bf16* __restrict__ src,
                                                     int ldb, int k0, int n0,
                                                     int lane) {
  int n  = n0 + (lane & 15);
  int ks = k0 + ((lane >> 4) << 3);
  const __bf16* p = src + (size_t)n * ldb + ks;
  v8bf lo = *(const v8bf*)(p);
  v8bf hi = *(const v8bf*)(p + 16);
  return __builtin_shufflevector(lo, hi, 0, 1, 2, 3, 4, 5, 6, 7,
                                 8, 9, 10, 11, 12, 13, 14, 15);
}

// ---------------------------------------------------------------------------
// Problem constants
// ---------------------------------------------------------------------------
#define HH 16
#define DD 64
#define DMODEL 1024
#define LL 256
#define BB 64
#define MTOT (LL * BB)          // 16384
#define SCALE_V (0.125f)        // 1/sqrt(64)
#define LN_EPS_V 1e-5f

// ---------------------------------------------------------------------------
// fp32 -> bf16 convert
// ---------------------------------------------------------------------------
__global__ void f32_to_bf16_kernel(const float* __restrict__ in,
                                   __bf16* __restrict__ out, int n) {
  int i = blockIdx.x * 256 + threadIdx.x;
  if (i < n) out[i] = (__bf16)in[i];
}

// ---------------------------------------------------------------------------
// NT bf16 GEMM: C[M,N] = A[M,K] * B[N,K]^T  (fp32 accumulate/output)
// Block: 256 threads = 8 waves (2 x 4). Wave tile 32x64 -> block tile 64x256.
// Fragments stream straight from global into WMMA; prefetch runs 2 K-tiles
// ahead (global_prefetch_b8, speculative => safe past buffer end).
// ---------------------------------------------------------------------------
__global__ void __launch_bounds__(256)
gemm_nt_bf16_kernel(const __bf16* __restrict__ A, const __bf16* __restrict__ B,
                    float* __restrict__ C, int M, int N, int K) {
  const int lane = threadIdx.x & 31;
  const int wave = threadIdx.x >> 5;
  const int wm = wave >> 2;       // 0..1
  const int wn = wave & 3;        // 0..3
  const int row0 = blockIdx.y * 64 + wm * 32;
  const int col0 = blockIdx.x * 256 + wn * 64;

  v8f acc[2][4];
#pragma unroll
  for (int i = 0; i < 2; ++i)
#pragma unroll
    for (int j = 0; j < 4; ++j) acc[i][j] = vzero8();

  for (int kb = 0; kb < K; kb += 32) {
    // prefetch 2 K-tiles ahead for this wave's A rows and B cols
    __builtin_prefetch(A + (size_t)(row0 + (lane & 15)) * K + kb + 64, 0, 0);
    __builtin_prefetch(A + (size_t)(row0 + 16 + (lane & 15)) * K + kb + 64, 0, 0);
    __builtin_prefetch(B + (size_t)(col0 + (lane & 15)) * K + kb + 64, 0, 0);

    v16bf a0 = load_a_frag(A, K, row0, kb, lane);
    v16bf a1 = load_a_frag(A, K, row0 + 16, kb, lane);
#pragma unroll
    for (int j = 0; j < 4; ++j) {
      v16bf bf = load_bT_frag(B, K, kb, col0 + j * 16, lane);
      acc[0][j] = wmma_bf16(a0, bf, acc[0][j]);
      acc[1][j] = wmma_bf16(a1, bf, acc[1][j]);
    }
  }

  const int off8 = (lane >> 4) << 3;
#pragma unroll
  for (int i = 0; i < 2; ++i)
#pragma unroll
    for (int j = 0; j < 4; ++j) {
      int colc = col0 + j * 16 + (lane & 15);
#pragma unroll
      for (int vv = 0; vv < 8; ++vv) {
        int r = row0 + i * 16 + off8 + vv;
        C[(size_t)r * N + colc] = acc[i][j][vv];
      }
    }
}

// ---------------------------------------------------------------------------
// phi + head-split + transpose: qkv[M,4096] fp32 -> q,k1,k2,v [B,H,L,D] bf16
// One wave per (m, h). phi = (elu(x)+1) / sum_D(elu(x)+1)
// ---------------------------------------------------------------------------
__global__ void __launch_bounds__(256)
phi_split_kernel(const float* __restrict__ qkv, __bf16* __restrict__ qb,
                 __bf16* __restrict__ k1b, __bf16* __restrict__ k2b,
                 __bf16* __restrict__ vb) {
  int gid = blockIdx.x * 256 + threadIdx.x;
  int wid = gid >> 5;             // global wave id: 0 .. M*H-1
  int lane = gid & 31;
  int m = wid >> 4;               // row index (l*B + b)
  int h = wid & 15;
  int l = m >> 6;                 // B == 64
  int b = m & 63;

  const float* base = qkv + (size_t)m * (4 * HH * DD) + h * (4 * DD);
  size_t obase = (((size_t)(b * HH + h)) * LL + l) * DD;

  __bf16* outs[3] = {qb, k1b, k2b};
#pragma unroll
  for (int comp = 0; comp < 3; ++comp) {
    float x0 = base[comp * DD + lane];
    float x1 = base[comp * DD + 32 + lane];
    float e0 = (x0 > 0.f) ? (x0 + 1.f) : __expf(x0);
    float e1 = (x1 > 0.f) ? (x1 + 1.f) : __expf(x1);
    float s = e0 + e1;
#pragma unroll
    for (int o = 16; o >= 1; o >>= 1) s += __shfl_xor(s, o, 32);
    float inv = 1.f / s;
    outs[comp][obase + lane] = (__bf16)(e0 * inv);
    outs[comp][obase + 32 + lane] = (__bf16)(e1 * inv);
  }
  // v: plain convert
  vb[obase + lane] = (__bf16)base[3 * DD + lane];
  vb[obase + 32 + lane] = (__bf16)base[3 * DD + 32 + lane];
}

// ---------------------------------------------------------------------------
// Fast-weight kernel: one workgroup (4 waves) per (b,h).
// Per 64-chunk: out = (q@k^T masked)@v + q@W_prev  for both memories,
// gated mix -> mixed[L,B,H*D] bf16; then W += k^T v.
// Tiles staged with GLOBAL_LOAD_ASYNC_TO_LDS_B128 (ASYNCcnt) when available,
// then transposed LDS->LDS for the B-operand layout.
// W kept transposed (W^T[e][d]) in LDS fp32 + bf16 shadow for B-operand use.
// ---------------------------------------------------------------------------
__global__ void __launch_bounds__(128)
fastweight_kernel(const __bf16* __restrict__ qb, const __bf16* __restrict__ k1b,
                  const __bf16* __restrict__ k2b, const __bf16* __restrict__ vb,
                  const float* __restrict__ pi0, __bf16* __restrict__ mixed) {
  const int bh = blockIdx.x;           // b*16 + h
  const int b = bh >> 4;
  const int h = bh & 15;
  const int tid = threadIdx.x;
  const int lane = tid & 31;
  const int wave = tid >> 5;           // 0..3
  const int t0 = wave * 16;            // wave's row strip within chunk
  const int off8 = (lane >> 4) << 3;

  __shared__ __bf16 vT[64 * 80];       // v^T[d][s]
  __shared__ __bf16 kT1[64 * 80];      // k1^T[d][s]
  __shared__ __bf16 kT2[64 * 80];
  __shared__ __bf16 sc[64 * 80];       // masked scores bf16 [t][s]
  __shared__ float W1[64 * 72];        // W1^T[e][d] fp32
  __shared__ float W2[64 * 72];
  __shared__ __bf16 W1b[64 * 80];      // bf16 shadow of W^T
  __shared__ __bf16 W2b[64 * 80];
#if USE_ASYNC
  __shared__ __bf16 rowv[64 * 64];     // row-major async landing buffers
  __shared__ __bf16 rowk1[64 * 64];
  __shared__ __bf16 rowk2[64 * 64];
#endif

  for (int i = tid; i < 64 * 72; i += 128) { W1[i] = 0.f; W2[i] = 0.f; }
  for (int i = tid; i < 64 * 80; i += 128) {
    W1b[i] = (__bf16)0.f;
    W2b[i] = (__bf16)0.f;
  }
  __syncthreads();

  const size_t base = (size_t)bh * LL * DD;
  const __bf16* qp = qb + base;
  const __bf16* kp1 = k1b + base;
  const __bf16* kp2 = k2b + base;
  const __bf16* vp = vb + base;

  for (int c = 0; c < 4; ++c) {
    const int l0 = c * 64;
#if USE_ASYNC
    // async-stage row-major 64x64 bf16 tiles (16B per lane per op)
    for (int i = tid; i < 512; i += 128) {
      int s = i >> 3, d8 = (i & 7) << 3;
      size_t go = (size_t)(l0 + s) * DD + d8;
      int lo = s * 64 + d8;
      async_copy_b128(vp + go, rowv + lo);
      async_copy_b128(kp1 + go, rowk1 + lo);
      async_copy_b128(kp2 + go, rowk2 + lo);
    }
    __builtin_amdgcn_s_wait_asynccnt(0);
    __syncthreads();
    // LDS->LDS transpose into B-operand layout
    for (int i = tid; i < 4096; i += 128) {
      int s = i >> 6, d = i & 63;
      vT[d * 80 + s]  = rowv[s * 64 + d];
      kT1[d * 80 + s] = rowk1[s * 64 + d];
      kT2[d * 80 + s] = rowk2[s * 64 + d];
    }
#else
    // fallback: transposing copy straight from global
    for (int i = tid; i < 4096; i += 128) {
      int s = i >> 6, d = i & 63;
      vT[d * 80 + s]  = vp[(size_t)(l0 + s) * DD + d];
      kT1[d * 80 + s] = kp1[(size_t)(l0 + s) * DD + d];
      kT2[d * 80 + s] = kp2[(size_t)(l0 + s) * DD + d];
    }
#endif
    __syncthreads();

    // q A-fragments (from global, row-major, K=d)
    v16bf aq0 = load_a_frag(qp, DD, l0 + t0, 0, lane);
    v16bf aq1 = load_a_frag(qp, DD, l0 + t0, 32, lane);

    v8f acc1[4], acc2[4];
#pragma unroll
    for (int j = 0; j < 4; ++j) { acc1[j] = vzero8(); acc2[j] = vzero8(); }

    // inter-chunk: q @ W_prev (B operand = W^T bf16 shadow)
#pragma unroll
    for (int j = 0; j < 4; ++j) {
      acc1[j] = wmma_bf16(aq0, load_bT_frag(W1b, 80, 0, j * 16, lane), acc1[j]);
      acc1[j] = wmma_bf16(aq1, load_bT_frag(W1b, 80, 32, j * 16, lane), acc1[j]);
      acc2[j] = wmma_bf16(aq0, load_bT_frag(W2b, 80, 0, j * 16, lane), acc2[j]);
      acc2[j] = wmma_bf16(aq1, load_bT_frag(W2b, 80, 32, j * 16, lane), acc2[j]);
    }

    // ---- memory 1: masked scores then scores @ v ----
    {
      v8f s1[4];
#pragma unroll
      for (int j = 0; j < 4; ++j) {
        s1[j] = vzero8();
        s1[j] = wmma_bf16(aq0, load_bT_frag(kp1 + (size_t)l0 * DD, DD, 0, j * 16, lane), s1[j]);
        s1[j] = wmma_bf16(aq1, load_bT_frag(kp1 + (size_t)l0 * DD, DD, 32, j * 16, lane), s1[j]);
      }
#pragma unroll
      for (int j = 0; j < 4; ++j) {
        int scol = j * 16 + (lane & 15);
#pragma unroll
        for (int vv = 0; vv < 8; ++vv) {
          int t = t0 + off8 + vv;
          float val = (scol <= t) ? s1[j][vv] : 0.f;
          sc[t * 80 + scol] = (__bf16)val;
        }
      }
      __syncthreads();
      v16bf as0 = load_a_frag(sc, 80, t0, 0, lane);
      v16bf as1 = load_a_frag(sc, 80, t0, 32, lane);
#pragma unroll
      for (int j = 0; j < 4; ++j) {
        acc1[j] = wmma_bf16(as0, load_bT_frag(vT, 80, 0, j * 16, lane), acc1[j]);
        acc1[j] = wmma_bf16(as1, load_bT_frag(vT, 80, 32, j * 16, lane), acc1[j]);
      }
      __syncthreads();
    }

    // ---- memory 2 ----
    {
      v8f s2[4];
#pragma unroll
      for (int j = 0; j < 4; ++j) {
        s2[j] = vzero8();
        s2[j] = wmma_bf16(aq0, load_bT_frag(kp2 + (size_t)l0 * DD, DD, 0, j * 16, lane), s2[j]);
        s2[j] = wmma_bf16(aq1, load_bT_frag(kp2 + (size_t)l0 * DD, DD, 32, j * 16, lane), s2[j]);
      }
#pragma unroll
      for (int j = 0; j < 4; ++j) {
        int scol = j * 16 + (lane & 15);
#pragma unroll
        for (int vv = 0; vv < 8; ++vv) {
          int t = t0 + off8 + vv;
          float val = (scol <= t) ? s2[j][vv] : 0.f;
          sc[t * 80 + scol] = (__bf16)val;
        }
      }
      __syncthreads();
      v16bf as0 = load_a_frag(sc, 80, t0, 0, lane);
      v16bf as1 = load_a_frag(sc, 80, t0, 32, lane);
#pragma unroll
      for (int j = 0; j < 4; ++j) {
        acc2[j] = wmma_bf16(as0, load_bT_frag(vT, 80, 0, j * 16, lane), acc2[j]);
        acc2[j] = wmma_bf16(as1, load_bT_frag(vT, 80, 32, j * 16, lane), acc2[j]);
      }
      __syncthreads();
    }

    // ---- gate with pi0 and store mixed[L,B,H*D] bf16 ----
#pragma unroll
    for (int vv = 0; vv < 8; ++vv) {
      int t = t0 + off8 + vv;
      int lidx = l0 + t;
      float p = pi0[h * 256 + lidx];
      p = fminf(fmaxf(p, 0.f), 1.f);
#pragma unroll
      for (int j = 0; j < 4; ++j) {
        float o = SCALE_V * (acc1[j][vv] * p + acc2[j][vv] * (1.f - p));
        int colg = h * DD + j * 16 + (lane & 15);
        mixed[((size_t)lidx * BB + b) * (HH * DD) + colg] = (__bf16)o;
      }
    }

    // ---- W updates: W^T[e][d] += (k^T v)^T, wave owns d-strip [t0,t0+16) ----
    {
      v16bf ak0 = load_a_frag(kT1, 80, t0, 0, lane);
      v16bf ak1 = load_a_frag(kT1, 80, t0, 32, lane);
      v16bf bk0 = load_a_frag(kT2, 80, t0, 0, lane);
      v16bf bk1 = load_a_frag(kT2, 80, t0, 32, lane);
#pragma unroll
      for (int j = 0; j < 4; ++j) {
        int e = j * 16 + (lane & 15);
        v16bf bv0 = load_bT_frag(vT, 80, 0, j * 16, lane);
        v16bf bv1 = load_bT_frag(vT, 80, 32, j * 16, lane);

        float* wp1 = W1 + e * 72 + t0 + off8;
        v8f cc1 = *(v8f*)wp1;
        cc1 = wmma_bf16(ak0, bv0, cc1);
        cc1 = wmma_bf16(ak1, bv1, cc1);
        *(v8f*)wp1 = cc1;

        float* wp2 = W2 + e * 72 + t0 + off8;
        v8f cc2 = *(v8f*)wp2;
        cc2 = wmma_bf16(bk0, bv0, cc2);
        cc2 = wmma_bf16(bk1, bv1, cc2);
        *(v8f*)wp2 = cc2;
      }
    }
    __syncthreads();
    // refresh bf16 shadow of W^T for next chunk's q@W
    for (int i = tid; i < 4096; i += 128) {
      int e = i >> 6, d = i & 63;
      W1b[e * 80 + d] = (__bf16)W1[e * 72 + d];
      W2b[e * 80 + d] = (__bf16)W2[e * 72 + d];
    }
    __syncthreads();
  }
}

// ---------------------------------------------------------------------------
// residual + post-LayerNorm: out = LN(h + attn) * gamma + beta
// ---------------------------------------------------------------------------
__global__ void __launch_bounds__(256)
ln_residual_kernel(const float* __restrict__ hin, const float* __restrict__ attn,
                   const float* __restrict__ gamma, const float* __restrict__ beta,
                   float* __restrict__ out) {
  int m = blockIdx.x;
  int tid = threadIdx.x;
  __shared__ float red[256];

  float x[4];
  float s = 0.f, s2 = 0.f;
#pragma unroll
  for (int i = 0; i < 4; ++i) {
    int col = tid + i * 256;
    float v = hin[(size_t)m * DMODEL + col] + attn[(size_t)m * DMODEL + col];
    x[i] = v;
    s += v;
    s2 += v * v;
  }
  red[tid] = s;
  __syncthreads();
  for (int st = 128; st > 0; st >>= 1) {
    if (tid < st) red[tid] += red[tid + st];
    __syncthreads();
  }
  float mu = red[0] * (1.f / DMODEL);
  __syncthreads();
  red[tid] = s2;
  __syncthreads();
  for (int st = 128; st > 0; st >>= 1) {
    if (tid < st) red[tid] += red[tid + st];
    __syncthreads();
  }
  float var = red[0] * (1.f / DMODEL) - mu * mu;
  float rstd = rsqrtf(var + LN_EPS_V);
#pragma unroll
  for (int i = 0; i < 4; ++i) {
    int col = tid + i * 256;
    out[(size_t)m * DMODEL + col] = (x[i] - mu) * rstd * gamma[col] + beta[col];
  }
}

// ---------------------------------------------------------------------------
// Workspace layout (bytes); qkv_f32 region reused as attn_f32, h_bf as mixed.
// ---------------------------------------------------------------------------
static const size_t OFF_QKV  = 0;                                   // 256 MB f32 (later attn)
static const size_t OFF_HBF  = (size_t)MTOT * 4096 * 4;             // 32 MB bf16 (later mixed)
static const size_t OFF_QKVW = OFF_HBF + (size_t)MTOT * DMODEL * 2; // 8 MB bf16
static const size_t OFF_OW   = OFF_QKVW + (size_t)4096 * DMODEL * 2;
static const size_t OFF_QB   = OFF_OW + (size_t)DMODEL * DMODEL * 2;
static const size_t SZ_HEAD  = (size_t)BB * HH * LL * DD * 2;       // 32 MB each
static const size_t OFF_K1B  = OFF_QB + SZ_HEAD;
static const size_t OFF_K2B  = OFF_K1B + SZ_HEAD;
static const size_t OFF_VB   = OFF_K2B + SZ_HEAD;

extern "C" void kernel_launch(void* const* d_in, const int* in_sizes, int n_in,
                              void* d_out, int out_size, void* d_ws, size_t ws_size,
                              hipStream_t stream) {
  (void)in_sizes; (void)n_in; (void)out_size; (void)ws_size;
  const float* h      = (const float*)d_in[0];
  const float* qkv_w  = (const float*)d_in[1];
  const float* o_w    = (const float*)d_in[2];
  const float* gammap = (const float*)d_in[3];
  const float* betap  = (const float*)d_in[4];
  const float* pi0    = (const float*)d_in[5];

  char* ws = (char*)d_ws;
  float*  qkv_f32  = (float*)(ws + OFF_QKV);
  float*  attn_f32 = (float*)(ws + OFF_QKV);   // alias: qkv consumed before reuse
  __bf16* h_bf     = (__bf16*)(ws + OFF_HBF);
  __bf16* mixed_bf = (__bf16*)(ws + OFF_HBF);  // alias: h_bf consumed before reuse
  __bf16* qkvw_bf  = (__bf16*)(ws + OFF_QKVW);
  __bf16* ow_bf    = (__bf16*)(ws + OFF_OW);
  __bf16* qb       = (__bf16*)(ws + OFF_QB);
  __bf16* k1b      = (__bf16*)(ws + OFF_K1B);
  __bf16* k2b      = (__bf16*)(ws + OFF_K2B);
  __bf16* vb       = (__bf16*)(ws + OFF_VB);

  // 1) converts
  {
    int n0 = MTOT * DMODEL;        // 16.7M
    int n1 = 4096 * DMODEL;        // 4.2M
    int n2 = DMODEL * DMODEL;      // 1.0M
    f32_to_bf16_kernel<<<(n0 + 255) / 256, 256, 0, stream>>>(h, h_bf, n0);
    f32_to_bf16_kernel<<<(n1 + 255) / 256, 256, 0, stream>>>(qkv_w, qkvw_bf, n1);
    f32_to_bf16_kernel<<<(n2 + 255) / 256, 256, 0, stream>>>(o_w, ow_bf, n2);
  }

  // 2) QKV projection: [16384,4096] = h_bf[16384,1024] @ qkv_w[4096,1024]^T
  gemm_nt_bf16_kernel<<<dim3(4096 / 256, MTOT / 64), 256, 0, stream>>>(
      h_bf, qkvw_bf, qkv_f32, MTOT, 4096, DMODEL);

  // 3) phi + split/transpose to [B,H,L,D]
  phi_split_kernel<<<(MTOT * HH * 32) / 256, 256, 0, stream>>>(qkv_f32, qb, k1b,
                                                               k2b, vb);

  // 4) dual fast-weight attention + pi0 gate -> mixed[L,B,H*D] bf16
  fastweight_kernel<<<BB * HH, 128, 0, stream>>>(qb, k1b, k2b, vb, pi0, mixed_bf);

  // 5) output projection: attn[16384,1024] = mixed @ o_w[1024,1024]^T
  gemm_nt_bf16_kernel<<<dim3(DMODEL / 256, MTOT / 64), 256, 0, stream>>>(
      mixed_bf, ow_bf, attn_f32, MTOT, DMODEL, DMODEL);

  // 6) residual + post-LN
  ln_residual_kernel<<<MTOT, 256, 0, stream>>>(h, attn_f32, gammap, betap,
                                               (float*)d_out);
}
